// SimpleConvolutionGNN_1357209666175
// MI455X (gfx1250) — matrix operand order repaired and checked by reading, not compiled, not run on hardware
//
#include <hip/hip_runtime.h>
#include <hip/hip_bf16.h>
#include <stdint.h>

#define HOPS 2
#define H 64
#define NUM_GRAPHS 16
#define ETPW 8   // edge tiles per wave
#define NTPW 4   // node tiles per wave

typedef __attribute__((ext_vector_type(16))) __bf16 v16bf;
typedef __attribute__((ext_vector_type(8)))  float  v8f;

struct Frag32 { uint4 lo, hi; };  // 32 bytes == v16bf

static __device__ __forceinline__ unsigned short f2bf(float f) {
  unsigned u = __float_as_uint(f);
  u += 0x7fffu + ((u >> 16) & 1u);   // round-to-nearest-even
  return (unsigned short)(u >> 16);
}

static __device__ __forceinline__ v16bf ld_frag(const unsigned short* p0,
                                                const unsigned short* p1) {
  Frag32 t;
  t.lo = *(const uint4*)p0;
  t.hi = *(const uint4*)p1;
  return __builtin_bit_cast(v16bf, t);
}

#define WMMA_BF16(a, b, c) \
  __builtin_amdgcn_wmma_f32_16x16x32_bf16(false, (a), false, (b), (short)0, (c), false, false)

// Pack W [HOPS][128][H] (f32, row-major) into per-lane WMMA-B bf16 layout:
// Wp[hop][kc][nt][lane][j]; lane<16: K=kc*32+j, N=nt*16+lane
//                           lane>=16: K=kc*32+16+j, N=nt*16+lane-16
__global__ void pack_w_kernel(const float* __restrict__ W,
                              unsigned short* __restrict__ Wp, int total) {
  int idx = blockIdx.x * blockDim.x + threadIdx.x;
  if (idx >= total) return;
  const int j    = idx & 15;
  const int lane = (idx >> 4) & 31;
  const int nt   = (idx >> 9) & 3;
  const int kc   = (idx >> 11) & 3;
  const int hop  = idx >> 13;
  const int r  = lane & 15;
  const int hl = lane >> 4;
  const int k = kc * 32 + hl * 16 + j;
  const int n = nt * 16 + r;
  Wp[idx] = f2bf(W[((long)hop * 128 + k) * H + n]);
}

__global__ void cvt_bf16_kernel(const float* __restrict__ src,
                                unsigned short* __restrict__ dst, long n) {
  long i = (long)blockIdx.x * blockDim.x + threadIdx.x;
  if (i < n) dst[i] = f2bf(src[i]);
}

// One wave processes ETPW consecutive 16-edge tiles, holding the full packed
// weight matrix (16 B-fragments = 128 VGPRs) in registers across tiles:
// msg = relu([h[src]; e_emb] @ Wmsg + b), atomically scattered to pooled[target].
__global__ __launch_bounds__(128) void edge_msg_kernel(
    const unsigned short* __restrict__ hbf,
    const unsigned short* __restrict__ ebf,
    const unsigned short* __restrict__ wp,    // this hop: [4][4][32][16] bf16
    const float* __restrict__ bias,           // this hop: [64]
    const int* __restrict__ esrc,
    const int* __restrict__ etgt,
    float* __restrict__ pooled,
    int ntiles) {
  const int lane = threadIdx.x & 31;
  const int wgl  = blockIdx.x * 4 + (threadIdx.x >> 5);
  const int r  = lane & 15;
  const int hl = lane >> 4;

  v16bf wf[16];                               // weights resident in VGPRs
#pragma unroll
  for (int i = 0; i < 16; ++i) {
    const unsigned short* w = wp + i * 512 + lane * 16;
    wf[i] = ld_frag(w, w + 8);
  }
  float bias4[4];
#pragma unroll
  for (int nt = 0; nt < 4; ++nt) bias4[nt] = bias[nt * 16 + r];

  const int t0 = wgl * ETPW;
  for (int t = 0; t < ETPW; ++t) {
    const int tile = t0 + t;
    if (tile >= ntiles) return;               // wave-uniform exit
    const long ebase = (long)tile * 16;
    const int src = esrc[ebase + r];

    v8f c0 = {0.f,0.f,0.f,0.f,0.f,0.f,0.f,0.f};
    v8f c1 = c0, c2 = c0, c3 = c0;
#pragma unroll
    for (int kc = 0; kc < 4; ++kc) {
      // A rows: K 0..63 from h[src], K 64..127 from edge_embedding[e]
      const unsigned short* arow = (kc < 2) ? (hbf + (long)src * H)
                                            : (ebf + (ebase + r) * H);
      const int koff = (kc & 1) * 32 + hl * 16;
      v16bf a = ld_frag(arow + koff, arow + koff + 8);
      c0 = WMMA_BF16(a, wf[kc * 4 + 0], c0);
      c1 = WMMA_BF16(a, wf[kc * 4 + 1], c1);
      c2 = WMMA_BF16(a, wf[kc * 4 + 2], c2);
      c3 = WMMA_BF16(a, wf[kc * 4 + 3], c3);
    }
    int tgt[8];
#pragma unroll
    for (int j = 0; j < 8; ++j) tgt[j] = etgt[ebase + hl * 8 + j];
    v8f acc[4] = {c0, c1, c2, c3};
#pragma unroll
    for (int nt = 0; nt < 4; ++nt) {
      const int col = nt * 16 + r;
#pragma unroll
      for (int j = 0; j < 8; ++j) {           // C layout: VGPR j -> row j + 8*hl
        float v = acc[nt][j] + bias4[nt];
        v = v > 0.f ? v : 0.f;
        atomicAdd(pooled + (long)tgt[j] * H + col, v);
      }
    }
  }
}

// One wave per 16-node tile (NTPW tiles per wave), weights in registers:
// h = LN(relu(h + [h; pooled] @ Wres + b)). Wave-local LDS staging for LN,
// ordered with s_wait_dscnt (LDS ops are in-order per wave; slice is private).
__global__ __launch_bounds__(64) void node_res_ln_kernel(
    const unsigned short* __restrict__ pbf,
    const unsigned short* __restrict__ wp,
    const float* __restrict__ bias,
    const float* __restrict__ gamma,
    const float* __restrict__ beta,
    float* __restrict__ hf,
    unsigned short* __restrict__ hbf,
    int ntiles) {
  __shared__ float lds[2][16][H];
  const int lane = threadIdx.x & 31;
  const int wave = threadIdx.x >> 5;
  const int wgl  = blockIdx.x * 2 + wave;
  const int r  = lane & 15;
  const int hl = lane >> 4;

  v16bf wf[16];
#pragma unroll
  for (int i = 0; i < 16; ++i) {
    const unsigned short* w = wp + i * 512 + lane * 16;
    wf[i] = ld_frag(w, w + 8);
  }
  float bias4[4];
#pragma unroll
  for (int nt = 0; nt < 4; ++nt) bias4[nt] = bias[nt * 16 + r];

  const int t0 = wgl * NTPW;
  for (int t = 0; t < NTPW; ++t) {
    const int tile = t0 + t;
    if (tile >= ntiles) return;               // wave-uniform exit
    const long nbase = (long)tile * 16;

    v8f c0 = {0.f,0.f,0.f,0.f,0.f,0.f,0.f,0.f};
    v8f c1 = c0, c2 = c0, c3 = c0;
#pragma unroll
    for (int kc = 0; kc < 4; ++kc) {
      const unsigned short* arow =
          ((kc < 2) ? (const unsigned short*)hbf : pbf) + (nbase + r) * H;
      const int koff = (kc & 1) * 32 + hl * 16;
      v16bf a = ld_frag(arow + koff, arow + koff + 8);
      c0 = WMMA_BF16(a, wf[kc * 4 + 0], c0);
      c1 = WMMA_BF16(a, wf[kc * 4 + 1], c1);
      c2 = WMMA_BF16(a, wf[kc * 4 + 2], c2);
      c3 = WMMA_BF16(a, wf[kc * 4 + 3], c3);
    }
    v8f acc[4] = {c0, c1, c2, c3};
#pragma unroll
    for (int nt = 0; nt < 4; ++nt) {
      const int col = nt * 16 + r;
#pragma unroll
      for (int j = 0; j < 8; ++j) {
        const int row = hl * 8 + j;
        float v = hf[(nbase + row) * H + col] + acc[nt][j] + bias4[nt];
        lds[wave][row][col] = v > 0.f ? v : 0.f;
      }
    }
    // Wave-local fence: all ds_stores above complete before ds_loads below.
    asm volatile("s_wait_dscnt 0x0" ::: "memory");

    float mean = 0.f;
#pragma unroll
    for (int c = 0; c < H; ++c) mean += lds[wave][r][c];
    mean *= (1.0f / H);
    float var = 0.f;
#pragma unroll
    for (int c = 0; c < H; ++c) { float d = lds[wave][r][c] - mean; var += d * d; }
    var *= (1.0f / H);
    const float inv = rsqrtf(var + 1e-3f);
    const long node = nbase + r;
#pragma unroll
    for (int k = 0; k < 32; ++k) {            // lane L: cols 0..31, lane L+16: cols 32..63
      const int col = hl * 32 + k;
      const float y = (lds[wave][r][col] - mean) * inv * gamma[col] + beta[col];
      hf[node * H + col] = y;
      hbf[node * H + col] = f2bf(y);
    }
    // LDS in-order per wave: next iteration's stores cannot pass these loads.
  }
}

__global__ void pool_accum_kernel(const float* __restrict__ hf,
                                  const int* __restrict__ gids,
                                  float* __restrict__ ctx,
                                  float* __restrict__ cnt, long total) {
  long i = (long)blockIdx.x * blockDim.x + threadIdx.x;
  if (i >= total) return;
  const long node = i >> 6;
  const int col = (int)(i & 63);
  const int g = gids[node];
  atomicAdd(ctx + g * H + col, hf[i]);
  if (col == 0) atomicAdd(cnt + g, 1.0f);
}

__global__ void pool_final_kernel(const float* __restrict__ ctx,
                                  const float* __restrict__ cnt,
                                  float* __restrict__ out) {
  int i = blockIdx.x * blockDim.x + threadIdx.x;
  if (i >= NUM_GRAPHS * H) return;
  float c = cnt[i >> 6];
  c = c < 1.f ? 1.f : c;
  out[i] = ctx[i] / c;
}

extern "C" void kernel_launch(void* const* d_in, const int* in_sizes, int n_in,
                              void* d_out, int out_size, void* d_ws, size_t ws_size,
                              hipStream_t stream) {
  const float* hidden = (const float*)d_in[0];
  const float* eemb   = (const float*)d_in[1];
  const int*   esrc   = (const int*)d_in[2];
  const int*   etgt   = (const int*)d_in[3];
  const int*   gids   = (const int*)d_in[4];
  const float* Wmsg   = (const float*)d_in[5];
  const float* bmsg   = (const float*)d_in[6];
  const float* Wres   = (const float*)d_in[7];
  const float* bres   = (const float*)d_in[8];
  const float* gamma  = (const float*)d_in[9];
  const float* beta   = (const float*)d_in[10];
  (void)n_in; (void)ws_size; (void)out_size;

  const long N = in_sizes[0] / H;
  const long E = in_sizes[2];

  char* ws = (char*)d_ws;
  auto alignup = [](size_t x) { return (x + 255) & ~(size_t)255; };
  size_t off = 0;
  unsigned short* hbf = (unsigned short*)(ws + off); off = alignup(off + (size_t)N * H * 2);
  unsigned short* ebf = (unsigned short*)(ws + off); off = alignup(off + (size_t)E * H * 2);
  float* pooledf      = (float*)(ws + off);          off = alignup(off + (size_t)N * H * 4);
  unsigned short* pbf = (unsigned short*)(ws + off); off = alignup(off + (size_t)N * H * 2);
  float* hf           = (float*)(ws + off);          off = alignup(off + (size_t)N * H * 4);
  unsigned short* wmsg_p = (unsigned short*)(ws + off); off = alignup(off + (size_t)HOPS * 128 * H * 2);
  unsigned short* wres_p = (unsigned short*)(ws + off); off = alignup(off + (size_t)HOPS * 128 * H * 2);
  float* ctx          = (float*)(ws + off);          off = alignup(off + (size_t)NUM_GRAPHS * H * 4);
  float* cnt          = (float*)(ws + off);

  const int packTotal = HOPS * 128 * H;
  pack_w_kernel<<<(packTotal + 255) / 256, 256, 0, stream>>>(Wmsg, wmsg_p, packTotal);
  pack_w_kernel<<<(packTotal + 255) / 256, 256, 0, stream>>>(Wres, wres_p, packTotal);
  cvt_bf16_kernel<<<(int)((N * H + 255) / 256), 256, 0, stream>>>(hidden, hbf, N * H);
  cvt_bf16_kernel<<<(int)((E * H + 255) / 256), 256, 0, stream>>>(eemb, ebf, E * H);
  hipMemcpyAsync(hf, hidden, (size_t)N * H * 4, hipMemcpyDeviceToDevice, stream);

  const int etiles = (int)((E + 15) / 16);
  const int ntilesN = (int)((N + 15) / 16);
  const int ewaves = (etiles + ETPW - 1) / ETPW;
  const int nwaves = (ntilesN + NTPW - 1) / NTPW;
  for (int hop = 0; hop < HOPS; ++hop) {
    hipMemsetAsync(pooledf, 0, (size_t)N * H * 4, stream);
    edge_msg_kernel<<<(ewaves + 3) / 4, 128, 0, stream>>>(
        hbf, ebf, wmsg_p + (size_t)hop * 128 * H, bmsg + hop * H,
        esrc, etgt, pooledf, etiles);
    cvt_bf16_kernel<<<(int)((N * H + 255) / 256), 256, 0, stream>>>(pooledf, pbf, N * H);
    node_res_ln_kernel<<<(nwaves + 1) / 2, 64, 0, stream>>>(
        pbf, wres_p + (size_t)hop * 128 * H, bres + hop * H,
        gamma, beta, hf, hbf, ntilesN);
  }

  hipMemsetAsync(ctx, 0, (size_t)NUM_GRAPHS * H * 4, stream);
  hipMemsetAsync(cnt, 0, (size_t)NUM_GRAPHS * 4, stream);
  pool_accum_kernel<<<(int)((N * H + 255) / 256), 256, 0, stream>>>(hf, gids, ctx, cnt, N * H);
  pool_final_kernel<<<(NUM_GRAPHS * H + 255) / 256, 256, 0, stream>>>(ctx, cnt, (float*)d_out);
}
